// NERLSTM_63513976373379
// MI455X (gfx1250) — matrix operand order repaired and checked by reading, not compiled, not run on hardware
//
#include <hip/hip_runtime.h>
#include <hip/hip_bf16.h>
#include <stdint.h>

// ---------------- types ----------------
typedef __attribute__((ext_vector_type(16))) __bf16 v16bf;
typedef __attribute__((ext_vector_type(8)))  float  v8f;

union FragB16 {
  v16bf v;
  unsigned u[8];
  uint4 q[2];
  unsigned short s[16];
};

#define BATCH 128
#define SEQT  512
#define FEATD 300
#define EDIM  256
#define GDIM  1024
#define NROWS 65536   // B*T
#define ROWT  4096    // NROWS/16
#define NEGV  -10000.0f

// ---------------- helpers ----------------
static __device__ __forceinline__ unsigned short f2bf(float f) {
  union { float f; unsigned u; } c; c.f = f;
  unsigned u = c.u;
  u += 0x7FFFu + ((u >> 16) & 1u);              // round-to-nearest-even
  return (unsigned short)(u >> 16);
}
static __device__ __forceinline__ float sigf(float x) {
  return 1.0f / (1.0f + __expf(-x));
}
static __device__ __forceinline__ float tanh_f(float x) {
  return 2.0f / (1.0f + __expf(-2.0f * x)) - 1.0f;
}
// A-matrix 16x32 bf16 per-lane dword -> K index (ISA 7.12.2)
static __device__ __forceinline__ int a_klo(int v, int g) {
  return ((v < 4) ? 0 : 16) + g * 8 + 2 * (v & 3);
}
// B-matrix 32x16 bf16 per-lane dword -> K index
static __device__ __forceinline__ int b_klo(int v, int g) {
  return ((v < 4) ? 0 : 16) + g * 8 + 2 * (v & 3);
}
// load a pre-packed fragment (1024B per fragment, 32B per lane, contiguous)
static __device__ __forceinline__ v16bf load_pk_frag(const unsigned short* base, int frag) {
  int lane = threadIdx.x & 31;
  const uint4* p = (const uint4*)(base + (size_t)frag * 512 + lane * 16);
  FragB16 f; f.q[0] = p[0]; f.q[1] = p[1];
  return f.v;
}

// ---------------- pack x into WMMA A-fragment layout (pad FEAT 300->320) ----------------
__global__ __launch_bounds__(256) void pack_x_kernel(const float* __restrict__ x,
                                                     unsigned short* __restrict__ dst) {
  int frag = blockIdx.x * 8 + (threadIdx.x >> 5);
  if (frag >= ROWT * 10) return;
  int lane = threadIdx.x & 31, g = lane >> 4, m = lane & 15;
  int rt = frag / 10, kt = frag % 10;
  int row = rt * 16 + m;
  FragB16 f;
#pragma unroll
  for (int v = 0; v < 8; ++v) {
    int k = kt * 32 + a_klo(v, g);
    float a0 = (k < FEATD)     ? x[(size_t)row * FEATD + k]     : 0.f;
    float a1 = (k + 1 < FEATD) ? x[(size_t)row * FEATD + k + 1] : 0.f;
    f.u[v] = (unsigned)f2bf(a0) | ((unsigned)f2bf(a1) << 16);
  }
  uint4* p = (uint4*)(dst + (size_t)frag * 512 + lane * 16);
  p[0] = f.q[0]; p[1] = f.q[1];
}

// ---------------- generic f32 weight -> packed bf16 B-fragments ----------------
// src: row-major [K][ldn]; rows >= split come from src1 (for concat(Wi,Wh)).
__global__ __launch_bounds__(256) void pack_b_kernel(const float* __restrict__ src0,
    const float* __restrict__ src1, int ldn, int kvalid, int nvalid, int split,
    int ntiles, int nfrags, unsigned short* __restrict__ dst) {
  int frag = blockIdx.x * 8 + (threadIdx.x >> 5);
  if (frag >= nfrags) return;
  int lane = threadIdx.x & 31, g = lane >> 4, l15 = lane & 15;
  int kt = frag / ntiles, nt = frag % ntiles;
  int ncol = nt * 16 + l15;
  FragB16 f;
#pragma unroll
  for (int v = 0; v < 8; ++v) {
    int k = kt * 32 + b_klo(v, g);
    float a0 = 0.f, a1 = 0.f;
    if (ncol < nvalid) {
      if (k < kvalid) {
        const float* sp = (k < split) ? (src0 + (size_t)k * ldn)
                                      : (src1 + (size_t)(k - split) * ldn);
        a0 = sp[ncol];
      }
      if (k + 1 < kvalid) {
        int k1 = k + 1;
        const float* sp = (k1 < split) ? (src0 + (size_t)k1 * ldn)
                                       : (src1 + (size_t)(k1 - split) * ldn);
        a1 = sp[ncol];
      }
    }
    f.u[v] = (unsigned)f2bf(a0) | ((unsigned)f2bf(a1) << 16);
  }
  uint4* p = (uint4*)(dst + (size_t)frag * 512 + lane * 16);
  p[0] = f.q[0]; p[1] = f.q[1];
}

// ---------------- embedding GEMM: emb = x @ Wemb + bemb (bf16 out) ----------------
__global__ __launch_bounds__(128) void gemm_emb_kernel(const unsigned short* __restrict__ xpk,
    const unsigned short* __restrict__ wpk, const float* __restrict__ bemb,
    unsigned short* __restrict__ emb) {
  __shared__ __align__(16) unsigned short S[16][256];
  int rt = blockIdx.x;
  int w = threadIdx.x >> 5, lane = threadIdx.x & 31, g = lane >> 4, l15 = lane & 15;
  v8f z = {0.f,0.f,0.f,0.f,0.f,0.f,0.f,0.f};
  v8f acc[4]; acc[0] = z; acc[1] = z; acc[2] = z; acc[3] = z;
  for (int kt = 0; kt < 10; ++kt) {
    v16bf a = load_pk_frag(xpk, rt * 10 + kt);
#pragma unroll
    for (int j = 0; j < 4; ++j) {
      v16bf b = load_pk_frag(wpk, kt * 16 + w * 4 + j);
      acc[j] = __builtin_amdgcn_wmma_f32_16x16x32_bf16(false, a, false, b,
                                                       (short)0, acc[j], false, false);
    }
  }
#pragma unroll
  for (int j = 0; j < 4; ++j) {
    int col = w * 64 + j * 16 + l15;
    float bb = bemb[col];
#pragma unroll
    for (int r = 0; r < 8; ++r)
      S[r + 8 * g][col] = f2bf(acc[j][r] + bb);
  }
  __syncthreads();
  // coalesced b128 writeback of the 16x256 bf16 tile
#pragma unroll
  for (int i = 0; i < 4; ++i) {
    int c = threadIdx.x + 128 * i;       // 0..511 8-element chunks
    int row = c >> 5, ccol = (c & 31) * 8;
    uint4 v4 = *(const uint4*)(&S[row][ccol]);
    *(uint4*)(emb + ((size_t)(rt * 16 + row)) * EDIM + ccol) = v4;
  }
}

// ---------------- persistent bidirectional LSTM recurrence ----------------
// grid (8 batch slices, 2 directions), 16 waves/WG. Wave w owns h-cols [16w,16w+16):
// gate tiles {w, w+16, w+32, w+48} -> cell update fully in-register.
// B fragments double-buffered across kt; h written back coalesced one step late,
// overlapped with the next step's GEMM.
__global__ __launch_bounds__(512) void lstm_kernel(const unsigned short* __restrict__ emb,
    const unsigned short* __restrict__ wcatpk, const float* __restrict__ bias_f,
    const float* __restrict__ bias_b, unsigned short* __restrict__ hf,
    unsigned short* __restrict__ hb) {
  __shared__ __align__(16) unsigned short A[16][520];   // [m][k], k<256 emb, k>=256 h
  const int tid = threadIdx.x, lane = tid & 31, w = tid >> 5;
  const int g = lane >> 4, l15 = lane & 15;
  const int dir = blockIdx.y;
  const int m0 = blockIdx.x * 16;
  const unsigned short* wpk = wcatpk + (size_t)dir * 1024 * 512;  // 1024 frags/dir
  const float* bias = dir ? bias_b : bias_f;
  unsigned short* hdst = dir ? hb : hf;

  const int n = w * 16 + l15;                 // this lane's h/gate column
  const float bi = bias[n], bfg = bias[n + 256], bgg = bias[n + 512], bog = bias[n + 768];

  const int rowm = tid >> 5;                  // 0..15 (loader/writeback mapping)
  const int cgrp = tid & 31;                  // 8 cols each
  { uint4 z4 = {0, 0, 0, 0}; *(uint4*)(&A[rowm][256 + cgrp * 8]) = z4; }  // h0 = 0
  float creg[8];
#pragma unroll
  for (int r = 0; r < 8; ++r) creg[r] = 0.f;  // c0 = 0

  { // preload emb for first step
    int tt = dir ? (SEQT - 1) : 0;
    const uint4* src = (const uint4*)(emb + ((size_t)(m0 + rowm) * SEQT + tt) * EDIM + cgrp * 8);
    *(uint4*)(&A[rowm][cgrp * 8]) = *src;
  }
  __syncthreads();

  const v8f z = {0.f,0.f,0.f,0.f,0.f,0.f,0.f,0.f};
  for (int s = 0; s < SEQT; ++s) {
    const int tt = dir ? (SEQT - 1 - s) : s;

    // coalesced writeback of previous step's h-tile (read-read overlap with GEMM)
    if (s > 0) {
      int tp = dir ? (SEQT - s) : (s - 1);
      uint4 hv = *(const uint4*)(&A[rowm][256 + cgrp * 8]);
      *(uint4*)(hdst + ((size_t)(m0 + rowm) * SEQT + tp) * EDIM + cgrp * 8) = hv;
    }

    v8f a0 = z, a1 = z, a2 = z, a3 = z;
    // software-pipelined B fragments (weights, L2-resident)
    v16bf b0 = load_pk_frag(wpk, w);
    v16bf b1 = load_pk_frag(wpk, 16 + w);
    v16bf b2 = load_pk_frag(wpk, 32 + w);
    v16bf b3 = load_pk_frag(wpk, 48 + w);
#pragma unroll 4
    for (int kt = 0; kt < 16; ++kt) {
      v16bf n0 = b0, n1 = b1, n2 = b2, n3 = b3;
      if (kt < 15) {
        n0 = load_pk_frag(wpk, (kt + 1) * 64 + w);
        n1 = load_pk_frag(wpk, (kt + 1) * 64 + 16 + w);
        n2 = load_pk_frag(wpk, (kt + 1) * 64 + 32 + w);
        n3 = load_pk_frag(wpk, (kt + 1) * 64 + 48 + w);
      }
      if (kt < 14)
        __builtin_prefetch(wpk + (size_t)((kt + 2) * 64 + w) * 512 + lane * 16, 0, 1);
      FragB16 fa;
#pragma unroll
      for (int v = 0; v < 8; ++v) {
        int k = kt * 32 + a_klo(v, g);
        fa.u[v] = *(const unsigned*)(&A[l15][k]);
      }
      a0 = __builtin_amdgcn_wmma_f32_16x16x32_bf16(false, fa.v, false, b0, (short)0, a0, false, false);
      a1 = __builtin_amdgcn_wmma_f32_16x16x32_bf16(false, fa.v, false, b1, (short)0, a1, false, false);
      a2 = __builtin_amdgcn_wmma_f32_16x16x32_bf16(false, fa.v, false, b2, (short)0, a2, false, false);
      a3 = __builtin_amdgcn_wmma_f32_16x16x32_bf16(false, fa.v, false, b3, (short)0, a3, false, false);
      b0 = n0; b1 = n1; b2 = n2; b3 = n3;
    }
    __syncthreads();                          // all A reads complete

    if (s + 1 < SEQT) {                       // stage next emb_t
      int tn = dir ? (SEQT - 2 - s) : (s + 1);
      const uint4* src = (const uint4*)(emb + ((size_t)(m0 + rowm) * SEQT + tn) * EDIM + cgrp * 8);
      *(uint4*)(&A[rowm][cgrp * 8]) = *src;
    }
#pragma unroll
    for (int r = 0; r < 8; ++r) {             // in-register LSTM cell (gate order i,f,g,o)
      float iv = sigf(a0[r] + bi);
      float fv = sigf(a1[r] + bfg);
      float gv = tanh_f(a2[r] + bgg);
      float ov = sigf(a3[r] + bog);
      float cn = fv * creg[r] + iv * gv;
      float hn = ov * tanh_f(cn);
      creg[r] = cn;
      const int M = r + 8 * g;
      A[M][256 + n] = f2bf(hn);               // feed next step's A
    }
    __syncthreads();
  }
  { // final step's h writeback
    int tp = dir ? 0 : (SEQT - 1);
    uint4 hv = *(const uint4*)(&A[rowm][256 + cgrp * 8]);
    *(uint4*)(hdst + ((size_t)(m0 + rowm) * SEQT + tp) * EDIM + cgrp * 8) = hv;
  }
}

// ---------------- output projection: feats = [h_f|h_b] @ Wout + bout ----------------
__global__ __launch_bounds__(32) void outproj_kernel(const unsigned short* __restrict__ hf,
    const unsigned short* __restrict__ hb, const unsigned short* __restrict__ wpk,
    const float* __restrict__ bout, float* __restrict__ feats) {
  __shared__ __align__(16) float SF[16][16];
  int rt = blockIdx.x;
  int lane = threadIdx.x & 31, g = lane >> 4, l15 = lane & 15;
  v8f acc = {0.f,0.f,0.f,0.f,0.f,0.f,0.f,0.f};
  for (int kt = 0; kt < 16; ++kt) {
    FragB16 fa;
    size_t row = (size_t)rt * 16 + l15;
#pragma unroll
    for (int v = 0; v < 8; ++v) {
      int k = kt * 32 + a_klo(v, g);
      const unsigned short* sp = (k < 256) ? (hf + row * EDIM + k)
                                           : (hb + row * EDIM + (k - 256));
      fa.u[v] = *(const unsigned*)sp;
    }
    v16bf b = load_pk_frag(wpk, kt);
    acc = __builtin_amdgcn_wmma_f32_16x16x32_bf16(false, fa.v, false, b, (short)0, acc, false, false);
  }
  float bb = (l15 < 11) ? bout[l15] : 0.f;
#pragma unroll
  for (int r = 0; r < 8; ++r)
    SF[r + 8 * g][l15] = acc[r] + bb;
  __syncthreads();
#pragma unroll
  for (int i = 0; i < 2; ++i) {
    int c = threadIdx.x + 32 * i;        // 0..63 4-float chunks
    int row = c >> 2, ccol = (c & 3) * 4;
    uint4 v4 = *(const uint4*)(&SF[row][ccol]);
    *(uint4*)(feats + ((size_t)(rt * 16 + row)) * 16 + ccol) = v4;
  }
}

// ---------------- Viterbi decode (one wave per batch element) ----------------
__global__ __launch_bounds__(32) void viterbi_kernel(const float* __restrict__ feats,
    const float* __restrict__ trans, int* __restrict__ out) {
  __shared__ float fvb[16];
  __shared__ float tmb[16];
  __shared__ unsigned char bp[SEQT][16];
  const int b = blockIdx.x, lane = threadIdx.x;
  float tr[11];
  if (lane < 11) {
#pragma unroll
    for (int p = 0; p < 11; ++p) tr[p] = trans[lane * 11 + p];   // trans[next][prev]
  }
  if (lane < 16) fvb[lane] = (lane == 9) ? 0.f : NEGV;           // START_IDX = 9
  __syncthreads();
  for (int t = 0; t < SEQT; ++t) {
    float best = -3.4e38f; int arg = 0;
    if (lane < 11) {
#pragma unroll
      for (int p = 0; p < 11; ++p) {
        float sc = fvb[p] + tr[p];
        if (sc > best) { best = sc; arg = p; }                   // first-max == argmax
      }
      best += feats[((size_t)b * SEQT + t) * 16 + lane];
    }
    __syncthreads();
    if (lane < 11) { fvb[lane] = best; bp[t][lane] = (unsigned char)arg; }
    __syncthreads();
  }
  if (lane < 16)
    tmb[lane] = (lane < 11) ? (fvb[lane] + trans[10 * 11 + lane]) : -3.4e38f;  // STOP_IDX=10
  __syncthreads();
  if (lane == 0) {
    int tag = 0; float bbest = tmb[0];
#pragma unroll
    for (int p = 1; p < 11; ++p)
      if (tmb[p] > bbest) { bbest = tmb[p]; tag = p; }
    for (int t = SEQT - 1; t >= 0; --t) {
      out[b * SEQT + t] = tag;
      tag = (int)bp[t][tag];
    }
  }
}

// ---------------- host launch ----------------
extern "C" void kernel_launch(void* const* d_in, const int* in_sizes, int n_in,
                              void* d_out, int out_size, void* d_ws, size_t ws_size,
                              hipStream_t stream) {
  const float* x     = (const float*)d_in[0];
  const float* Wemb  = (const float*)d_in[1];
  const float* bemb  = (const float*)d_in[2];
  const float* Wi_f  = (const float*)d_in[3];
  const float* Wh_f  = (const float*)d_in[4];
  const float* b_f   = (const float*)d_in[5];
  const float* Wi_b  = (const float*)d_in[6];
  const float* Wh_b  = (const float*)d_in[7];
  const float* b_b   = (const float*)d_in[8];
  const float* Wout  = (const float*)d_in[9];
  const float* bout  = (const float*)d_in[10];
  const float* trans = (const float*)d_in[11];
  int* out = (int*)d_out;

  char* ws = (char*)d_ws;
  unsigned short* xpk    = (unsigned short*)(ws + 0ull);           // 40 MB: x A-frags
  unsigned short* wembpk = (unsigned short*)(ws + 41943040ull);    // 160 KB
  unsigned short* emb    = (unsigned short*)(ws + 42106880ull);    // 32 MB bf16 [65536][256]
  unsigned short* wcatpk = (unsigned short*)(ws + 75661312ull);    // 2 MB: concat(Wi,Wh) x2 dirs
  unsigned short* woutpk = (unsigned short*)(ws + 77758464ull);    // 16 KB
  unsigned short* hf     = (unsigned short*)(ws + 77774848ull);    // 32 MB
  unsigned short* hb     = (unsigned short*)(ws + 111329280ull);   // 32 MB
  float*          feats  = (float*)        (ws + 144883712ull);    // 4 MB [65536][16]

  pack_x_kernel<<<5120, 256, 0, stream>>>(x, xpk);
  pack_b_kernel<<<20, 256, 0, stream>>>(Wemb, Wemb, 256, 300, 256, 1 << 30, 16, 160, wembpk);
  pack_b_kernel<<<128, 256, 0, stream>>>(Wi_f, Wh_f, 1024, 512, 1024, 256, 64, 1024, wcatpk);
  pack_b_kernel<<<128, 256, 0, stream>>>(Wi_b, Wh_b, 1024, 512, 1024, 256, 64, 1024,
                                         wcatpk + 1024ull * 512);
  pack_b_kernel<<<2, 256, 0, stream>>>(Wout, Wout, 11, 512, 11, 1 << 30, 1, 16, woutpk);

  gemm_emb_kernel<<<4096, 128, 0, stream>>>(xpk, wembpk, bemb, emb);
  lstm_kernel<<<dim3(8, 2), 512, 0, stream>>>(emb, wcatpk, b_f, b_b, hf, hb);
  outproj_kernel<<<4096, 32, 0, stream>>>(hf, hb, woutpk, bout, feats);
  viterbi_kernel<<<128, 32, 0, stream>>>(feats, trans, out);

  (void)in_sizes; (void)n_in; (void)out_size; (void)ws_size;
}
